// MoELayer_86002425135540
// MI455X (gfx1250) — matrix operand order repaired and checked by reading, not compiled, not run on hardware
//
#include <hip/hip_runtime.h>
#include <math.h>

// Problem dims (fixed by the reference)
#define M_TOK 2048   // B*S
#define HDIM  1024
#define EXP   8
#define FDIM  4096
#define GATE_TEMP 0.1f

typedef __attribute__((ext_vector_type(16))) __bf16 v16bf;
typedef __attribute__((ext_vector_type(8)))  float  v8f;
typedef __attribute__((ext_vector_type(8)))  unsigned short ush8;

union ABFrag { v16bf v; ush8 h[2]; };

// fp32 -> bf16, round to nearest even
__device__ __forceinline__ unsigned short f2b(float f) {
    unsigned int u = __float_as_uint(f);
    u += 0x7FFFu + ((u >> 16) & 1u);
    return (unsigned short)(u >> 16);
}

// ---- gfx1250 async global->LDS (ASYNCcnt-tracked), via inline asm ----
__device__ __forceinline__ void async_b128(unsigned lds_off, const void* gaddr) {
    asm volatile("global_load_async_to_lds_b128 %0, %1, off"
                 :: "v"(lds_off), "v"(gaddr) : "memory");
}
template <int N>
__device__ __forceinline__ void wait_async() {
    asm volatile("s_wait_asynccnt %0" :: "i"(N) : "memory");
}

// ---------------- convert fp32 -> bf16 (grid-stride) ----------------
__global__ void f2bf_kernel(const float* __restrict__ in,
                            unsigned short* __restrict__ out, size_t n) {
    size_t i = (size_t)blockIdx.x * blockDim.x + threadIdx.x;
    size_t stride = (size_t)gridDim.x * blockDim.x;
    for (; i < n; i += stride) out[i] = f2b(in[i]);
}

// ------- convert fp32 [Z,K,N] -> bf16 transposed [Z,N,K] (LDS-tiled) -------
__global__ __launch_bounds__(256) void transpose_f2bf_kernel(
    const float* __restrict__ in, unsigned short* __restrict__ out, int K, int N) {
    __shared__ float tile[32][33];
    const size_t z = (size_t)blockIdx.z * K * N;
    const int kBase = blockIdx.y * 32, nBase = blockIdx.x * 32;
    const int r = threadIdx.x >> 5, c = threadIdx.x & 31;
#pragma unroll
    for (int i = 0; i < 4; ++i)
        tile[r + 8 * i][c] = in[z + (size_t)(kBase + r + 8 * i) * N + nBase + c];
    __syncthreads();
#pragma unroll
    for (int i = 0; i < 4; ++i)
        out[z + (size_t)(nBase + r + 8 * i) * K + kBase + c] = f2b(tile[c][r + 8 * i]);
}

// ---------------- router / gating -> normalized combine weights ----------------
__global__ __launch_bounds__(256) void router_kernel(
    const float* __restrict__ x,   // [M,H]
    const float* __restrict__ wr,  // [H,E]
    const float* __restrict__ wg,  // [H,E]
    float* __restrict__ cw)        // [M,E]
{
    __shared__ float red[256 * 16];
    const int t = blockIdx.x;
    const int tid = threadIdx.x;

    float r[EXP], g[EXP];
#pragma unroll
    for (int e = 0; e < EXP; ++e) { r[e] = 0.f; g[e] = 0.f; }

    for (int i = tid; i < HDIM; i += 256) {
        float xv = x[(size_t)t * HDIM + i];
#pragma unroll
        for (int e = 0; e < EXP; ++e) {
            r[e] += xv * wr[i * EXP + e];
            g[e] += xv * wg[i * EXP + e];
        }
    }
#pragma unroll
    for (int e = 0; e < EXP; ++e) {
        red[tid * 16 + e]     = r[e];
        red[tid * 16 + 8 + e] = g[e];
    }
    __syncthreads();
    for (int s = 128; s > 0; s >>= 1) {
        if (tid < s) {
#pragma unroll
            for (int j = 0; j < 16; ++j)
                red[tid * 16 + j] += red[(tid + s) * 16 + j];
        }
        __syncthreads();
    }
    if (tid == 0) {
        float rl[EXP], gl[EXP];
#pragma unroll
        for (int e = 0; e < EXP; ++e) { rl[e] = red[e]; gl[e] = red[8 + e] / GATE_TEMP; }
        float rm = -1e30f, gm = -1e30f;
#pragma unroll
        for (int e = 0; e < EXP; ++e) { rm = fmaxf(rm, rl[e]); gm = fmaxf(gm, gl[e]); }
        float rs = 0.f, gs = 0.f, pr[EXP], pg[EXP];
#pragma unroll
        for (int e = 0; e < EXP; ++e) {
            pr[e] = __expf(rl[e] - rm); rs += pr[e];
            pg[e] = __expf(gl[e] - gm); gs += pg[e];
        }
        float c[EXP], cs = 0.f;
#pragma unroll
        for (int e = 0; e < EXP; ++e) { c[e] = (pr[e] / rs) * (pg[e] / gs); cs += c[e]; }
        float inv = 1.f / (cs + 1e-9f);
#pragma unroll
        for (int e = 0; e < EXP; ++e) cw[(size_t)t * EXP + e] = c[e] * inv;
    }
}

// ---------------- WMMA fragment loads from LDS ----------------
// K-tile is 64 wide per staged slice; LDA ushort stride 72 (144 B, multiple of 16 B,
// not a power of two -> staggers banks).
#define LDA 72

__device__ __forceinline__ v16bf load_a_frag(const unsigned short* lds, int row, int ks,
                                             int halfsel) {
    // 16-bit A 16x32: lanes 0-15 hold K 0..7 & 16..23; lanes 16-31 hold 8..15 & 24..31
    ABFrag f;
    const unsigned short* p = lds + row * LDA + ks * 32 + halfsel * 8;
    f.h[0] = *(const ush8*)p;
    f.h[1] = *(const ush8*)(p + 16);
    return f.v;
}
__device__ __forceinline__ v16bf load_b_frag(const unsigned short* lds, int col, int ks,
                                             int halfsel) {
    // 16-bit B 32x16: lanes 0-15 hold K 0..7 & 16..23 of col N=lane; lanes 16-31 hold 8..15 & 24..31
    ABFrag f;
    const unsigned short* p = lds + col * LDA + ks * 32 + halfsel * 8;
    f.h[0] = *(const ush8*)p;
    f.h[1] = *(const ush8*)(p + 16);
    return f.v;
}

__device__ __forceinline__ unsigned lds_off(const void* p) {
    return (unsigned)(size_t)p;  // LDS aperture keeps the offset in the low 32 bits
}

// ---------------- GEMM1: h' = cw[m,e] * silu(x Wg) * (x Wu), bf16 out ----------------
// grid: (FDIM/64, M_TOK/128, EXP), block 256 (8 wave32)
__global__ __launch_bounds__(256) void gemm1_kernel(
    const unsigned short* __restrict__ xb,   // [M,H] bf16
    const unsigned short* __restrict__ wgT,  // [E,F,H] bf16 (transposed)
    const unsigned short* __restrict__ wuT,  // [E,F,H] bf16 (transposed)
    const float* __restrict__ cw,            // [M,E]
    unsigned short* __restrict__ hbuf)       // [E,M,F] bf16
{
    __shared__ __align__(16) unsigned short xs[2][128 * LDA];
    __shared__ __align__(16) unsigned short gs[2][64 * LDA];
    __shared__ __align__(16) unsigned short us[2][64 * LDA];

    const int tid   = threadIdx.x;
    const int e     = blockIdx.z;
    const int mBase = blockIdx.y * 128;
    const int nBase = blockIdx.x * 64;

    const int wave    = tid >> 5;
    const int lane    = tid & 31;
    const int halfsel = lane >> 4;
    const int lrow    = lane & 15;
    const int waveM   = (wave & 3) * 32;
    const int waveN   = (wave >> 2) * 32;

    v8f accg[2][2], accu[2][2];
#pragma unroll
    for (int i = 0; i < 2; ++i)
#pragma unroll
        for (int j = 0; j < 2; ++j) {
            accg[i][j] = (v8f){0.f,0.f,0.f,0.f,0.f,0.f,0.f,0.f};
            accu[i][j] = (v8f){0.f,0.f,0.f,0.f,0.f,0.f,0.f,0.f};
        }

    // staging roles: all transfers are contiguous b128
    const int sxRow = tid >> 1, sxK = (tid & 1) * 32;  // X tile 128x64: 4 b128/thread
    const int swN   = tid >> 2, swK = (tid & 3) * 16;  // W tiles 64x64:  2 b128/thread

    const unsigned short* xsrc = xb  + (size_t)(mBase + sxRow) * HDIM + sxK;
    const unsigned short* gsrc = wgT + ((size_t)e * FDIM + nBase + swN) * HDIM + swK;
    const unsigned short* usrc = wuT + ((size_t)e * FDIM + nBase + swN) * HDIM + swK;

    auto stage = [&](int buf, int kt) {  // 8 async instructions per wave
        unsigned xo = lds_off(&xs[buf][sxRow * LDA + sxK]);
        async_b128(xo,      xsrc + kt);
        async_b128(xo + 16, xsrc + kt + 8);
        async_b128(xo + 32, xsrc + kt + 16);
        async_b128(xo + 48, xsrc + kt + 24);
        unsigned go = lds_off(&gs[buf][swN * LDA + swK]);
        async_b128(go,      gsrc + kt);
        async_b128(go + 16, gsrc + kt + 8);
        unsigned uo = lds_off(&us[buf][swN * LDA + swK]);
        async_b128(uo,      usrc + kt);
        async_b128(uo + 16, usrc + kt + 8);
    };

    stage(0, 0);
    const int NT = HDIM / 64;
    for (int it = 0; it < NT; ++it) {
        const int cur = it & 1;
        if (it + 1 < NT) { stage(cur ^ 1, (it + 1) * 64); wait_async<8>(); }
        else             { wait_async<0>(); }
        __syncthreads();

#pragma unroll
        for (int ks = 0; ks < 2; ++ks) {
            v16bf a0 = load_a_frag(xs[cur], waveM + lrow, ks, halfsel);
            v16bf a1 = load_a_frag(xs[cur], waveM + 16 + lrow, ks, halfsel);
#pragma unroll
            for (int ni = 0; ni < 2; ++ni) {
                v16bf bg = load_b_frag(gs[cur], waveN + ni * 16 + lrow, ks, halfsel);
                v16bf bu = load_b_frag(us[cur], waveN + ni * 16 + lrow, ks, halfsel);
                accg[0][ni] = __builtin_amdgcn_wmma_f32_16x16x32_bf16(false, a0, false, bg, (short)0, accg[0][ni], false, false);
                accg[1][ni] = __builtin_amdgcn_wmma_f32_16x16x32_bf16(false, a1, false, bg, (short)0, accg[1][ni], false, false);
                accu[0][ni] = __builtin_amdgcn_wmma_f32_16x16x32_bf16(false, a0, false, bu, (short)0, accu[0][ni], false, false);
                accu[1][ni] = __builtin_amdgcn_wmma_f32_16x16x32_bf16(false, a1, false, bu, (short)0, accu[1][ni], false, false);
            }
        }
        __syncthreads();
    }

    // epilogue: swiglu, fold in combine weight, store bf16
#pragma unroll
    for (int mi = 0; mi < 2; ++mi) {
        const int mrow = mBase + waveM + mi * 16 + 8 * halfsel; // +r = global token
        float cwv[8];
#pragma unroll
        for (int r = 0; r < 8; ++r) cwv[r] = cw[(size_t)(mrow + r) * EXP + e];
#pragma unroll
        for (int ni = 0; ni < 2; ++ni) {
            const int n = nBase + waveN + ni * 16 + lrow;
#pragma unroll
            for (int r = 0; r < 8; ++r) {
                float g = accg[mi][ni][r];
                float u = accu[mi][ni][r];
                float h = (g / (1.0f + __expf(-g))) * u * cwv[r];
                hbuf[((size_t)e * M_TOK + mrow + r) * FDIM + n] = f2b(h);
            }
        }
    }
}

// ---------------- GEMM2: out = sum_e H'_e * Wd_e, fp32 out ----------------
// grid: (HDIM/64, M_TOK/128), block 256
__global__ __launch_bounds__(256) void gemm2_kernel(
    const unsigned short* __restrict__ hbuf, // [E,M,F] bf16
    const unsigned short* __restrict__ wdT,  // [E,H,F] bf16 (transposed)
    float* __restrict__ out)                 // [M,H] fp32
{
    __shared__ __align__(16) unsigned short as[2][128 * LDA];
    __shared__ __align__(16) unsigned short bs[2][64 * LDA];

    const int tid   = threadIdx.x;
    const int mBase = blockIdx.y * 128;
    const int nBase = blockIdx.x * 64;

    const int wave    = tid >> 5;
    const int lane    = tid & 31;
    const int halfsel = lane >> 4;
    const int lrow    = lane & 15;
    const int waveM   = (wave & 3) * 32;
    const int waveN   = (wave >> 2) * 32;

    v8f acc[2][2];
#pragma unroll
    for (int i = 0; i < 2; ++i)
#pragma unroll
        for (int j = 0; j < 2; ++j)
            acc[i][j] = (v8f){0.f,0.f,0.f,0.f,0.f,0.f,0.f,0.f};

    const int sxRow = tid >> 1, sxK = (tid & 1) * 32;
    const int swN   = tid >> 2, swK = (tid & 3) * 16;

    auto stage = [&](int buf, int it) {  // 6 async instructions per wave
        const int e  = it >> 6;           // FDIM/64 = 64 k-tiles per expert
        const int kt = (it & 63) * 64;
        const unsigned short* ga = hbuf + ((size_t)e * M_TOK + mBase + sxRow) * FDIM + kt + sxK;
        unsigned ao = lds_off(&as[buf][sxRow * LDA + sxK]);
        async_b128(ao,      ga);
        async_b128(ao + 16, ga + 8);
        async_b128(ao + 32, ga + 16);
        async_b128(ao + 48, ga + 24);
        const unsigned short* gb = wdT + ((size_t)e * HDIM + nBase + swN) * FDIM + kt + swK;
        unsigned bo = lds_off(&bs[buf][swN * LDA + swK]);
        async_b128(bo,      gb);
        async_b128(bo + 16, gb + 8);
    };

    const int NT = EXP * (FDIM / 64);  // 512 k-tiles
    stage(0, 0);
    for (int it = 0; it < NT; ++it) {
        const int cur = it & 1;
        if (it + 1 < NT) { stage(cur ^ 1, it + 1); wait_async<6>(); }
        else             { wait_async<0>(); }
        __syncthreads();

#pragma unroll
        for (int ks = 0; ks < 2; ++ks) {
            v16bf a0 = load_a_frag(as[cur], waveM + lrow, ks, halfsel);
            v16bf a1 = load_a_frag(as[cur], waveM + 16 + lrow, ks, halfsel);
#pragma unroll
            for (int ni = 0; ni < 2; ++ni) {
                v16bf b = load_b_frag(bs[cur], waveN + ni * 16 + lrow, ks, halfsel);
                acc[0][ni] = __builtin_amdgcn_wmma_f32_16x16x32_bf16(false, a0, false, b, (short)0, acc[0][ni], false, false);
                acc[1][ni] = __builtin_amdgcn_wmma_f32_16x16x32_bf16(false, a1, false, b, (short)0, acc[1][ni], false, false);
            }
        }
        __syncthreads();
    }

#pragma unroll
    for (int mi = 0; mi < 2; ++mi) {
        const int mrow = mBase + waveM + mi * 16 + 8 * halfsel;
#pragma unroll
        for (int ni = 0; ni < 2; ++ni) {
            const int n = nBase + waveN + ni * 16 + lrow;
#pragma unroll
            for (int r = 0; r < 8; ++r)
                out[(size_t)(mrow + r) * HDIM + n] = acc[mi][ni][r];
        }
    }
}

// ---------------- launch ----------------
extern "C" void kernel_launch(void* const* d_in, const int* in_sizes, int n_in,
                              void* d_out, int out_size, void* d_ws, size_t ws_size,
                              hipStream_t stream) {
    (void)in_sizes; (void)n_in; (void)out_size; (void)ws_size;
    const float* x     = (const float*)d_in[0];  // [2,1024,1024]
    const float* wr    = (const float*)d_in[1];  // [1024,8]
    const float* wgt   = (const float*)d_in[2];  // [1024,8]
    const float* wgate = (const float*)d_in[3];  // [8,1024,4096]
    const float* wup   = (const float*)d_in[4];  // [8,1024,4096]
    const float* wdown = (const float*)d_in[5];  // [8,4096,1024]
    float* out = (float*)d_out;

    // workspace carve-up (~324.3 MB total)
    char* ws = (char*)d_ws;
    size_t off = 0;
    auto alloc = [&](size_t bytes) -> void* {
        void* p = ws + off;
        off += (bytes + 255) & ~(size_t)255;
        return p;
    };
    float*          cw   = (float*)         alloc((size_t)M_TOK * EXP * sizeof(float));
    unsigned short* xb   = (unsigned short*)alloc((size_t)M_TOK * HDIM * 2);
    unsigned short* wgT  = (unsigned short*)alloc((size_t)EXP * HDIM * FDIM * 2);
    unsigned short* wuT  = (unsigned short*)alloc((size_t)EXP * HDIM * FDIM * 2);
    unsigned short* wdT  = (unsigned short*)alloc((size_t)EXP * FDIM * HDIM * 2);
    unsigned short* hbuf = (unsigned short*)alloc((size_t)EXP * M_TOK * FDIM * 2);

    // 1) down-convert (and transpose weights to n-major) bf16
    f2bf_kernel<<<2048, 256, 0, stream>>>(x, xb, (size_t)M_TOK * HDIM);
    {   // wgate/wup: [E, H, F] -> [E, F, H]
        dim3 g(FDIM / 32, HDIM / 32, EXP);
        transpose_f2bf_kernel<<<g, 256, 0, stream>>>(wgate, wgT, HDIM, FDIM);
        transpose_f2bf_kernel<<<g, 256, 0, stream>>>(wup,   wuT, HDIM, FDIM);
    }
    {   // wdown: [E, F, H] -> [E, H, F]
        dim3 g(HDIM / 32, FDIM / 32, EXP);
        transpose_f2bf_kernel<<<g, 256, 0, stream>>>(wdown, wdT, FDIM, HDIM);
    }

    // 2) router+gating combine weights
    router_kernel<<<M_TOK, 256, 0, stream>>>(x, wr, wgt, cw);

    // 3) per-expert fused gate/up + swiglu + combine-weight fold
    dim3 g1(FDIM / 64, M_TOK / 128, EXP);
    gemm1_kernel<<<g1, 256, 0, stream>>>(xb, wgT, wuT, cw, hbuf);

    // 4) down-projection summed over experts
    dim3 g2(HDIM / 64, M_TOK / 128);
    gemm2_kernel<<<g2, 256, 0, stream>>>(hbuf, wdT, out);
}
